// GRU_52089363366078
// MI455X (gfx1250) — compile-verified, matrix operand order checked by reading, not run on hardware
//
#include <hip/hip_runtime.h>
#include <cstdint>
#include <cstddef>

// Problem dimensions (fixed by the reference).
#define BB 64
#define TT 1024
#define II 512
#define HH 512
#define NWG_REC 32   // persistent workgroups in the recurrent kernel (one per 16-wide N tile)

typedef __attribute__((ext_vector_type(16))) __bf16 v16bf;
typedef __attribute__((ext_vector_type(8)))  float  v8f;

union Frag16 {
    unsigned short s[16];
    uint4          q[2];
    v16bf          v;
};

// f32 -> bf16 via the hardware convert (RNE); avoids 3-op integer emulation.
__device__ __forceinline__ unsigned short f2bf(float f) {
    __bf16 h = (__bf16)f;
    return __builtin_bit_cast(unsigned short, h);
}

__device__ __forceinline__ float bf2f(unsigned short s) {
    union { unsigned u; float f; } x; x.u = ((unsigned)s) << 16;
    return x.f;
}

__device__ __forceinline__ float sigmoidf_(float x) {
    return 1.0f / (1.0f + __expf(-x));
}

__device__ __forceinline__ v8f wmma_bf16(v16bf a, v16bf b, v8f c) {
    // D = A(16x32 bf16) * B(32x16 bf16) + C(16x16 f32)
    return __builtin_amdgcn_wmma_f32_16x16x32_bf16(
        /*neg_a=*/false, a, /*neg_b=*/false, b,
        /*c_mod=*/(short)0, c, /*reuse_a=*/false, /*reuse_b=*/false);
}

// Load a 16x32 bf16 fragment from a row-major [rows x ld] bf16 (ushort) array.
// CDNA5 16-bit A layout: lane m = lane&15; lanes<16 hold K {0..7,16..23},
// lanes>=16 hold K {8..15,24..31}. B is supplied as W^T (row n of W == col n
// of B), so the identical loader serves both operands.
__device__ __forceinline__ v16bf load_frag_bf16(const unsigned short* base,
                                                int row0, int ld, int k0, int lane) {
    const int lo = (lane & 16) ? 8 : 0;
    const unsigned short* p = base + (size_t)(row0 + (lane & 15)) * ld + (k0 + lo);
    Frag16 f;
    f.q[0] = *(const uint4*)(p);
    f.q[1] = *(const uint4*)(p + 16);
    return f.v;
}

// ---------------------------------------------------------------------------
// Kernel 1: convert the six fp32 weight matrices to bf16, packed per gate.
// Wih = [r,z,n][H][I], Whh = [r,z,n][H][H]
// ---------------------------------------------------------------------------
__global__ void cvt_weights_kernel(const float* __restrict__ Wir, const float* __restrict__ Wiz,
                                   const float* __restrict__ Win, const float* __restrict__ Whr,
                                   const float* __restrict__ Whz, const float* __restrict__ Whn,
                                   unsigned short* __restrict__ Wih,
                                   unsigned short* __restrict__ Whh) {
    int i = blockIdx.x * blockDim.x + threadIdx.x;
    if (i >= HH * II) return;
    Wih[0 * HH * II + i] = f2bf(Wir[i]);
    Wih[1 * HH * II + i] = f2bf(Wiz[i]);
    Wih[2 * HH * II + i] = f2bf(Win[i]);
    Whh[0 * HH * HH + i] = f2bf(Whr[i]);
    Whh[1 * HH * HH + i] = f2bf(Whz[i]);
    Whh[2 * HH * HH + i] = f2bf(Whn[i]);
}

// ---------------------------------------------------------------------------
// Kernel 2: convert x (fp32, 128 MB) to bf16 once. 8 elements per thread,
// b128-in / b128-out. Removes all conversion VALU from the GEMM hot loop.
// ---------------------------------------------------------------------------
__global__ void cvt_x_kernel(const float* __restrict__ x, unsigned short* __restrict__ xbf) {
    const size_t i = ((size_t)blockIdx.x * blockDim.x + threadIdx.x) * 8;
    if (i >= (size_t)BB * TT * II) return;
    float4 a = *(const float4*)(x + i);
    float4 b = *(const float4*)(x + i + 4);
    union { unsigned short s[8]; uint4 q; } o;
    o.s[0] = f2bf(a.x); o.s[1] = f2bf(a.y); o.s[2] = f2bf(a.z); o.s[3] = f2bf(a.w);
    o.s[4] = f2bf(b.x); o.s[5] = f2bf(b.y); o.s[6] = f2bf(b.z); o.s[7] = f2bf(b.w);
    *(uint4*)(xbf + i) = o.q;
}

// ---------------------------------------------------------------------------
// Kernel 3: init h (f32 + bf16 copies) and the grid barrier state.
// Runs every launch -> graph-replay safe (no reliance on leftover ws state).
// ---------------------------------------------------------------------------
__global__ void init_kernel(const float* __restrict__ h0,
                            float* __restrict__ hf0, unsigned short* __restrict__ hb0,
                            unsigned* __restrict__ bar) {
    int i = blockIdx.x * blockDim.x + threadIdx.x;
    if (i < BB * HH) {
        float v = h0[i];
        hf0[i] = v;
        hb0[i] = f2bf(v);
    }
    if (i == 0) { bar[0] = 0u; bar[1] = 0u; }
}

// ---------------------------------------------------------------------------
// Kernel 4: input projections. xi[g][t][b][n] (bf16) = x[b,t,:] @ W_g^T + b_g
// One wave per 16x64 output strip: A-frag reused across 4 B-frags/4 WMMAs
// per k-step (~2.5 b128 loads per WMMA).
// ---------------------------------------------------------------------------
__global__ void __launch_bounds__(128) input_proj_kernel(
        const unsigned short* __restrict__ xbf,
        const unsigned short* __restrict__ Wih,
        const float* __restrict__ b_ir, const float* __restrict__ b_iz,
        const float* __restrict__ b_in,
        unsigned short* __restrict__ xi) {
    const int lane = threadIdx.x & 31;
    const int wave = blockIdx.x * (blockDim.x >> 5) + (threadIdx.x >> 5);
    // decompose flat wave id: nstrip(8, 64 wide) x mtile(4) x t(1024) x gate(3)
    const int ns = wave & 7;
    const int mt = (wave >> 3) & 3;
    const int t  = (wave >> 5) & 1023;
    const int g  = wave >> 15;
    if (g >= 3) return;

    const int n0 = ns * 64;
    const int b0 = mt * 16;
    const float* bg = (g == 0) ? b_ir : (g == 1) ? b_iz : b_in;

    v8f c[4];
    #pragma unroll
    for (int j = 0; j < 4; ++j) {
        const float bias = bg[n0 + j * 16 + (lane & 15)];
        #pragma unroll
        for (int v = 0; v < 8; ++v) c[j][v] = bias;
    }

    const unsigned short* Wg = Wih + (size_t)g * HH * II;
    const unsigned short* pa = xbf + ((size_t)(b0 + (lane & 15)) * TT + t) * II;

    for (int k0 = 0; k0 < II; k0 += 32) {
        v16bf a = load_frag_bf16(pa, 0, 0, k0, lane);   // row already folded into pa
        #pragma unroll
        for (int j = 0; j < 4; ++j) {
            v16bf b = load_frag_bf16(Wg, n0 + j * 16, II, k0, lane);
            c[j] = wmma_bf16(a, b, c[j]);
        }
    }

    unsigned short* po = xi + (((size_t)g * TT + t) * BB) * HH;
    const int hi8 = (lane & 16) ? 8 : 0;
    const int nl  = lane & 15;
    #pragma unroll
    for (int j = 0; j < 4; ++j) {
        #pragma unroll
        for (int v = 0; v < 8; ++v) {
            const int bidx = b0 + v + hi8;
            po[(size_t)bidx * HH + n0 + j * 16 + nl] = f2bf(c[j][v]);
        }
    }
}

// ---------------------------------------------------------------------------
// Kernel 5: persistent recurrent kernel.
// Grid = 32 WGs x 128 threads (4 waves). WG owns one 16-wide N slice; its
// W_hr/W_hz/W_hn rows are staged once into LDS with async global->LDS copies
// (ASYNCcnt path) and stay resident for all 1024 steps. Waves own 16-wide B
// slices. Per step: 16 k-steps x (1 A-frag from h_bf16, 3 LDS B-frags,
// 3 WMMAs), then fused gate math, h double-buffer flip, grid barrier.
// ---------------------------------------------------------------------------
__global__ void __launch_bounds__(128, 1) gru_recurrent_kernel(
        const unsigned short* __restrict__ xi,
        const unsigned short* __restrict__ Whh,
        const float* __restrict__ b_hn,
        float* __restrict__ hf0, float* __restrict__ hf1,
        unsigned short* __restrict__ hb0, unsigned short* __restrict__ hb1,
        float* __restrict__ out, float* __restrict__ hn_out,
        unsigned* __restrict__ bar) {
    __shared__ unsigned short ldsW[3 * 16 * HH];   // 48 KB: W_g[n0..n0+15, :] x 3 gates
    __shared__ float s_bhn[16];

    const int tid  = threadIdx.x;
    const int lane = tid & 31;
    const int wave = tid >> 5;            // m-tile (batch tile) 0..3
    const int n0   = blockIdx.x * 16;     // N slice owned by this WG
    const int b0   = wave * 16;

    // Stage this WG's weight slice into LDS with async global->LDS b128 copies.
    // Low 32 bits of a generic shared-memory address are the LDS byte offset
    // (ISA 10.2).
    {
        const unsigned short* src0 = Whh;
        for (int i = tid; i < 3 * 1024; i += blockDim.x) {     // 3072 x 16B = 48 KB
            const int g = i >> 10, j = i & 1023;
            const unsigned short* src = src0 + ((size_t)g * HH + n0) * HH + (size_t)j * 8;
            const unsigned ldsOff = (unsigned)(uintptr_t)&ldsW[((size_t)g * 1024 + j) * 8];
            asm volatile("global_load_async_to_lds_b128 %0, %1, off"
                         :: "v"(ldsOff), "v"((unsigned long long)(uintptr_t)src)
                         : "memory");
        }
        if (tid < 16) s_bhn[tid] = b_hn[n0 + tid];
        asm volatile("s_wait_asynccnt 0x0" ::: "memory");
    }
    __syncthreads();

    const int hi8 = (lane & 16) ? 8 : 0;
    const int nl  = lane & 15;
    float*          hf[2] = { hf0, hf1 };
    unsigned short* hb[2] = { hb0, hb1 };
    unsigned* bar_cnt = bar;
    unsigned* gen     = bar + 1;
    const float bhn = s_bhn[nl];

    for (int t = 0; t < TT; ++t) {
        const int cur = t & 1, nxt = cur ^ 1;
        const unsigned short* hcur = hb[cur];

        // Prefetch next step's xi lines for this wave's tile into cache.
        if (t + 1 < TT) {
            const size_t nx = ((size_t)(t + 1) * BB + b0) * HH + n0;
            __builtin_prefetch(xi + (size_t)0 * TT * BB * HH + nx, 0, 1);
            __builtin_prefetch(xi + (size_t)1 * TT * BB * HH + nx, 0, 1);
            __builtin_prefetch(xi + (size_t)2 * TT * BB * HH + nx, 0, 1);
        }

        v8f accR = {}, accZ = {}, accN = {};
        #pragma unroll 4
        for (int k0 = 0; k0 < HH; k0 += 32) {
            v16bf a  = load_frag_bf16(hcur, b0, HH, k0, lane);
            v16bf br = load_frag_bf16(ldsW + 0 * 16 * HH, 0, HH, k0, lane);
            v16bf bz = load_frag_bf16(ldsW + 1 * 16 * HH, 0, HH, k0, lane);
            v16bf bn = load_frag_bf16(ldsW + 2 * 16 * HH, 0, HH, k0, lane);
            accR = wmma_bf16(a, br, accR);
            accZ = wmma_bf16(a, bz, accZ);
            accN = wmma_bf16(a, bn, accN);
        }

        // Fused GRU gate math + h update + output store.
        #pragma unroll
        for (int v = 0; v < 8; ++v) {
            const int bidx = b0 + v + hi8;
            const int nidx = n0 + nl;
            const size_t idx = (size_t)bidx * HH + nidx;
            const size_t xio = ((size_t)t * BB + bidx) * HH + nidx;

            const float ir  = bf2f(xi[(size_t)0 * TT * BB * HH + xio]);
            const float iz  = bf2f(xi[(size_t)1 * TT * BB * HH + xio]);
            const float inn = bf2f(xi[(size_t)2 * TT * BB * HH + xio]);
            const float hv  = hf[cur][idx];

            const float r  = sigmoidf_(ir + accR[v]);
            const float z  = sigmoidf_(iz + accZ[v]);
            const float nn = tanhf(inn + r * (accN[v] + bhn));
            const float hn = (1.0f - z) * nn + z * hv;

            hf[nxt][idx] = hn;
            hb[nxt][idx] = f2bf(hn);
            out[((size_t)bidx * TT + t) * HH + nidx] = hn;
            if (t == TT - 1) hn_out[idx] = hn;
        }

        // Grid-wide barrier: monotonic generation counter keyed to t.
        __threadfence();
        __syncthreads();
        if (tid == 0) {
            const unsigned target = (unsigned)(t + 1);
            unsigned prev = __hip_atomic_fetch_add(bar_cnt, 1u, __ATOMIC_ACQ_REL,
                                                   __HIP_MEMORY_SCOPE_AGENT);
            if (prev == NWG_REC - 1) {
                __hip_atomic_store(bar_cnt, 0u, __ATOMIC_RELAXED,
                                   __HIP_MEMORY_SCOPE_AGENT);
                __hip_atomic_store(gen, target, __ATOMIC_RELEASE,
                                   __HIP_MEMORY_SCOPE_AGENT);
            } else {
                while (__hip_atomic_load(gen, __ATOMIC_ACQUIRE,
                                         __HIP_MEMORY_SCOPE_AGENT) < target) {
                    __builtin_amdgcn_s_sleep(1);
                }
            }
        }
        __syncthreads();
        __threadfence();
    }
}

// ---------------------------------------------------------------------------
// Host-side launcher.
// ---------------------------------------------------------------------------
extern "C" void kernel_launch(void* const* d_in, const int* in_sizes, int n_in,
                              void* d_out, int out_size, void* d_ws, size_t ws_size,
                              hipStream_t stream) {
    (void)in_sizes; (void)n_in; (void)out_size; (void)ws_size;

    const float* x   = (const float*)d_in[0];
    const float* h0  = (const float*)d_in[1];
    const float* Wir = (const float*)d_in[2];
    const float* bir = (const float*)d_in[3];
    const float* Whr = (const float*)d_in[4];
    const float* Wiz = (const float*)d_in[5];
    const float* biz = (const float*)d_in[6];
    const float* Whz = (const float*)d_in[7];
    const float* Win = (const float*)d_in[8];
    const float* bin = (const float*)d_in[9];
    const float* Whn = (const float*)d_in[10];
    const float* bhn = (const float*)d_in[11];
    float* out = (float*)d_out;

    // Workspace layout.
    char* ws = (char*)d_ws;
    size_t off = 0;
    unsigned* bar = (unsigned*)ws;                         off += 256;
    unsigned short* xi  = (unsigned short*)(ws + off);     off += (size_t)3 * TT * BB * HH * 2;
    unsigned short* xbf = (unsigned short*)(ws + off);     off += (size_t)BB * TT * II * 2;
    unsigned short* Wih = (unsigned short*)(ws + off);     off += (size_t)3 * HH * II * 2;
    unsigned short* Whh = (unsigned short*)(ws + off);     off += (size_t)3 * HH * HH * 2;
    float* hf0 = (float*)(ws + off);                       off += (size_t)BB * HH * 4;
    float* hf1 = (float*)(ws + off);                       off += (size_t)BB * HH * 4;
    unsigned short* hb0 = (unsigned short*)(ws + off);     off += (size_t)BB * HH * 2;
    unsigned short* hb1 = (unsigned short*)(ws + off);     off += (size_t)BB * HH * 2;

    cvt_weights_kernel<<<(HH * II + 255) / 256, 256, 0, stream>>>(
        Wir, Wiz, Win, Whr, Whz, Whn, Wih, Whh);

    cvt_x_kernel<<<(int)(((size_t)BB * TT * II / 8 + 255) / 256), 256, 0, stream>>>(x, xbf);

    init_kernel<<<(BB * HH + 255) / 256, 256, 0, stream>>>(h0, hf0, hb0, bar);

    // 3 gates x 1024 t x 4 m-tiles x 8 n-strips = 98304 waves, 4 waves/block.
    input_proj_kernel<<<24576, 128, 0, stream>>>(xbf, Wih, bir, biz, bin, xi);

    gru_recurrent_kernel<<<NWG_REC, 128, 0, stream>>>(
        xi, Whh, bhn, hf0, hf1, hb0, hb1,
        out, out + (size_t)BB * TT * HH, bar);
}
